// WindGatedDynamicGraphBuilder_11501922418699
// MI455X (gfx1250) — compile-verified, matrix operand order checked by reading, not run on hardware
//
#include <hip/hip_runtime.h>
#include <hip/hip_bf16.h>

// Shapes from the reference
#define NN   256   // N nodes
#define DM   64    // d_model
#define DQK  32    // d_qk
#define DEMB 40
#define GH   32
#define BL   384   // B*L = 16*24

typedef __attribute__((ext_vector_type(16))) __bf16 v16bf;
typedef __attribute__((ext_vector_type(4)))  __bf16 v4bf;
typedef __attribute__((ext_vector_type(8)))  float  v8f;

__device__ __forceinline__ float softplus_f(float x) {
  return (x > 20.f) ? x : log1pf(expf(x));
}

// ---------------------------------------------------------------------------
// Fragment (16x32 bf16, one 32-long K-vector per lane) from row-major LDS.
// ISA 16-bit A layout: lanes 0-15 hold K={0..7,16..23}, lanes 16-31 K={8..15,24..31}.
// Used for A (rows of the left matrix) and, with a transposed LDS image, for B
// (columns of the right matrix) — both are two contiguous 16B ds loads.
__device__ __forceinline__ v16bf load_frag(const __bf16* base, int stride,
                                           int r0, int k0, int ln, int half) {
  const __bf16* p = base + (r0 + ln) * stride + k0 + half * 8;
  v16bf a;
#pragma unroll
  for (int j = 0; j < 8; ++j) { a[j] = p[j]; a[j + 8] = p[16 + j]; }
  return a;
}

// ---------------------------------------------------------------------------
// Prep: P = a*A_static + b*softmax(ne·neT), invDen[i] = 1/(a*ΣA_static[i]+b+c+1+eps)
__global__ void __launch_bounds__(256)
prep_kernel(const float* __restrict__ Ast, const float* __restrict__ ne,
            const float* __restrict__ alphaP, const float* __restrict__ betaP,
            const float* __restrict__ gammaP,
            float* __restrict__ P, float* __restrict__ invDen)
{
  __shared__ float nrow[DEMB];
  __shared__ float redM[8], redE[8], redS[8];
  const int i = blockIdx.x;
  const int j = threadIdx.x;
  const int w = j >> 5, lane = j & 31;

  if (j < DEMB) nrow[j] = ne[i * DEMB + j];
  __syncthreads();

  float dot = 0.f;
  for (int d = 0; d < DEMB; ++d) dot += nrow[d] * ne[j * DEMB + d];

  float m = dot;
  for (int off = 1; off < 32; off <<= 1) m = fmaxf(m, __shfl_xor(m, off, 32));
  if (lane == 0) redM[w] = m;
  __syncthreads();
  m = redM[0];
  for (int q = 1; q < 8; ++q) m = fmaxf(m, redM[q]);

  float e  = expf(dot - m);
  float as = Ast[i * NN + j];

  float se = e, ss = as;
  for (int off = 1; off < 32; off <<= 1) {
    se += __shfl_xor(se, off, 32);
    ss += __shfl_xor(ss, off, 32);
  }
  if (lane == 0) { redE[w] = se; redS[w] = ss; }
  __syncthreads();
  se = 0.f; ss = 0.f;
  for (int q = 0; q < 8; ++q) { se += redE[q]; ss += redS[q]; }

  const float a = softplus_f(alphaP[0]);
  const float b = softplus_f(betaP[0]);
  const float c = softplus_f(gammaP[0]);

  P[i * NN + j] = a * as + b * (e / se);
  if (j == 0) invDen[i] = 1.0f / (a * ss + b + c + 1.0f + 1e-8f);
}

// ---------------------------------------------------------------------------
// Wind gate MLP: rscale[idx] = (1 + 0.5*sigmoid(mlp(uv))) / sqrt(DQK)
__global__ void __launch_bounds__(256)
gate_kernel(const float* __restrict__ wind,
            const float* __restrict__ w1, const float* __restrict__ b1,
            const float* __restrict__ w2, const float* __restrict__ b2,
            float* __restrict__ rscale)
{
  const int idx = blockIdx.x * 256 + threadIdx.x;   // BL*NN threads exactly
  const float u0 = wind[idx * 3 + 0];
  const float u1 = wind[idx * 3 + 1];
  const float u2 = wind[idx * 3 + 2];
  float s = b2[0];
#pragma unroll 4
  for (int g = 0; g < GH; ++g) {
    float x  = fmaf(u0, w1[g], fmaf(u1, w1[GH + g], fmaf(u2, w1[2 * GH + g], b1[g])));
    float gl = 0.5f * x * (1.f + erff(x * 0.70710678118f));   // exact-erf GELU
    s = fmaf(gl, w2[g], s);
  }
  const float gate = 1.f / (1.f + expf(-s));
  rscale[idx] = (1.f + 0.5f * gate) * 0.17677669529663687f;   // (1+λg)/sqrt(32)
}

// ---------------------------------------------------------------------------
// Main attention + fusion kernel. One workgroup per (b,l); 8 waves.
__global__ void __launch_bounds__(256)
attn_kernel(const float* __restrict__ h,
            const float* __restrict__ Wq, const float* __restrict__ Wk,
            const float* __restrict__ P,  const float* __restrict__ invDen,
            const float* __restrict__ rscale, const float* __restrict__ gammaP,
            float* __restrict__ out)
{
  __shared__ __bf16 lds[NN * DM];         // phase1: h (256x64). phase2: q | k (each 256x32)
  __shared__ __bf16 ldsWT[2 * DM * DQK];  // Wq^T, Wk^T (each 32x64, n-major)

  const int tid = threadIdx.x;
  const int bl  = blockIdx.x;
  const int w    = tid >> 5;
  const int lane = tid & 31;
  const int half = lane >> 4;
  const int ln   = lane & 15;

  const float* hbl = h + (size_t)bl * NN * DM;

  // Stage h as bf16: float4 global loads, packed 8B LDS stores
  {
    const float4* h4 = (const float4*)hbl;
    for (int i = tid; i < NN * DM / 4; i += 256) {
      float4 x = h4[i];
      v4bf y;
      y[0] = (__bf16)x.x; y[1] = (__bf16)x.y; y[2] = (__bf16)x.z; y[3] = (__bf16)x.w;
      *(v4bf*)&lds[i * 4] = y;
    }
  }
  // Stage W transposed so B-fragments are contiguous (scatter once here)
  for (int i = tid; i < DM * DQK; i += 256) {
    const int k = i >> 5;          // row of W   (0..63)
    const int n = i & (DQK - 1);   // col of W   (0..31)
    ldsWT[n * DM + k]            = (__bf16)Wq[i];
    ldsWT[DM * DQK + n * DM + k] = (__bf16)Wk[i];
  }
  __syncthreads();

  // ---- Phase 1: q = h*Wq, k = h*Wk ----
  // Preload all 8 W fragments (reused by both row-tiles of this wave).
  v16bf BW[2][2][2];   // [m][kc][ns]
#pragma unroll
  for (int m = 0; m < 2; ++m)
#pragma unroll
    for (int kc = 0; kc < 2; ++kc)
#pragma unroll
      for (int ns = 0; ns < 2; ++ns)
        BW[m][kc][ns] = load_frag(ldsWT + m * DM * DQK, DM, ns * 16, kc * 32, ln, half);

  v8f qk[2][4];   // [rowtile][m*2+nsub]
#pragma unroll
  for (int t = 0; t < 2; ++t) {
    const int r0 = (w + t * 8) * 16;
    const v16bf A0 = load_frag(lds, DM, r0, 0,  ln, half);
    const v16bf A1 = load_frag(lds, DM, r0, 32, ln, half);
#pragma unroll
    for (int m = 0; m < 2; ++m)
#pragma unroll
      for (int ns = 0; ns < 2; ++ns) {
        v8f acc = {};
        acc = __builtin_amdgcn_wmma_f32_16x16x32_bf16(false, A0, false, BW[m][0][ns],
                                                      (short)0, acc, false, false);
        acc = __builtin_amdgcn_wmma_f32_16x16x32_bf16(false, A1, false, BW[m][1][ns],
                                                      (short)0, acc, false, false);
        qk[t][m * 2 + ns] = acc;
      }
  }
  __syncthreads();   // everyone done reading h

  // write q (first 256x32) and k (second 256x32) over the h region, as bf16
#pragma unroll
  for (int t = 0; t < 2; ++t) {
    const int r0 = (w + t * 8) * 16;
#pragma unroll
    for (int m = 0; m < 2; ++m)
#pragma unroll
      for (int ns = 0; ns < 2; ++ns) {
        v8f acc = qk[t][m * 2 + ns];
#pragma unroll
        for (int v = 0; v < 8; ++v) {
          const int row = r0 + v + 8 * half;
          const int col = ns * 16 + ln;
          lds[m * (NN * DQK) + row * DQK + col] = (__bf16)acc[v];
        }
      }
  }
  __syncthreads();

  const float c = softplus_f(gammaP[0]);
  const __bf16* ldsQ = lds;
  const __bf16* ldsK = lds + NN * DQK;

  // ---- Phase 2: logits, softmax, fused epilogue ----
#pragma unroll 1
  for (int t = 0; t < 2; ++t) {
    const int r0 = (w + t * 8) * 16;
    const v16bf Aq = load_frag(ldsQ, DQK, r0, 0, ln, half);

    // rows held by this lane are r0+8*half .. r0+8*half+7 -> vector loads
    float rs[8], idv[8], mrow[8], srow[8];
    {
      const float4* r4 = (const float4*)(rscale + bl * NN + r0 + 8 * half);
      const float4* d4 = (const float4*)(invDen + r0 + 8 * half);
      float4 ra = r4[0], rb = r4[1], da = d4[0], db = d4[1];
      rs[0] = ra.x; rs[1] = ra.y; rs[2] = ra.z; rs[3] = ra.w;
      rs[4] = rb.x; rs[5] = rb.y; rs[6] = rb.z; rs[7] = rb.w;
      idv[0] = da.x; idv[1] = da.y; idv[2] = da.z; idv[3] = da.w;
      idv[4] = db.x; idv[5] = db.y; idv[6] = db.z; idv[7] = db.w;
    }
#pragma unroll
    for (int v = 0; v < 8; ++v) { mrow[v] = -3.0e38f; srow[v] = 0.f; }

    v8f acc[16];
#pragma unroll
    for (int jt = 0; jt < 16; ++jt) {
      v16bf Bk = load_frag(ldsK, DQK, jt * 16, 0, ln, half);  // B col n = k row n
      v8f z = {};
      acc[jt] = __builtin_amdgcn_wmma_f32_16x16x32_bf16(false, Aq, false, Bk,
                                                        (short)0, z, false, false);
    }

    // scale by per-row gate (includes 1/sqrt(dqk)) and find row max
#pragma unroll
    for (int jt = 0; jt < 16; ++jt)
#pragma unroll
      for (int v = 0; v < 8; ++v) {
        float x = acc[jt][v] * rs[v];
        acc[jt][v] = x;
        mrow[v] = fmaxf(mrow[v], x);
      }
#pragma unroll
    for (int v = 0; v < 8; ++v)
      for (int off = 1; off < 16; off <<= 1)
        mrow[v] = fmaxf(mrow[v], __shfl_xor(mrow[v], off, 16));

    // exp + row sum
#pragma unroll
    for (int jt = 0; jt < 16; ++jt)
#pragma unroll
      for (int v = 0; v < 8; ++v) {
        float e = __expf(acc[jt][v] - mrow[v]);
        acc[jt][v] = e;
        srow[v] += e;
      }
#pragma unroll
    for (int v = 0; v < 8; ++v) {
      for (int off = 1; off < 16; off <<= 1)
        srow[v] += __shfl_xor(srow[v], off, 16);
      srow[v] = 1.0f / srow[v];
    }

    // epilogue: (P + eye + c*A_dyn) * invDen, coalesced 64B ops per lane-half
#pragma unroll
    for (int jt = 0; jt < 16; ++jt) {
      const int col = jt * 16 + ln;
#pragma unroll
      for (int v = 0; v < 8; ++v) {
        const int row = r0 + v + 8 * half;
        float pv = P[row * NN + col]
                 + ((row == col) ? 1.0f : 0.0f)
                 + c * acc[jt][v] * srow[v];
        out[((size_t)bl * NN + row) * NN + col] = pv * idv[v];
      }
    }
  }
}

// ---------------------------------------------------------------------------
extern "C" void kernel_launch(void* const* d_in, const int* in_sizes, int n_in,
                              void* d_out, int out_size, void* d_ws, size_t ws_size,
                              hipStream_t stream) {
  const float* h     = (const float*)d_in[0];
  const float* Ast   = (const float*)d_in[1];
  const float* wind  = (const float*)d_in[2];
  const float* Wq    = (const float*)d_in[3];
  const float* Wk    = (const float*)d_in[4];
  const float* ne    = (const float*)d_in[5];
  const float* gw1   = (const float*)d_in[6];
  const float* gb1   = (const float*)d_in[7];
  const float* gw2   = (const float*)d_in[8];
  const float* gb2   = (const float*)d_in[9];
  const float* alpha = (const float*)d_in[10];
  const float* beta  = (const float*)d_in[11];
  const float* gamma = (const float*)d_in[12];
  float* out = (float*)d_out;

  float* wsf    = (float*)d_ws;
  float* P      = wsf;                    // 256*256
  float* invDen = wsf + NN * NN;          // 256
  float* rscale = wsf + NN * NN + NN;     // 384*256

  prep_kernel<<<NN, 256, 0, stream>>>(Ast, ne, alpha, beta, gamma, P, invDen);
  gate_kernel<<<BL, 256, 0, stream>>>(wind, gw1, gb1, gw2, gb2, rscale);
  attn_kernel<<<BL, 256, 0, stream>>>(h, Wq, Wk, P, invDen, rscale, gamma, out);
}
